// BahdanauAttention_12541304504858
// MI455X (gfx1250) — matrix operand order repaired and checked by reading, not compiled, not run on hardware
//
#include <hip/hip_runtime.h>
#include <math.h>

// Problem sizes (fixed by the reference)
#define BATCH 32
#define TLEN  8192
#define DENC  256
#define DATT  512
#define TTILE 128              // rows of encoder handled per block (4 waves * 32)
#define NCHUNK (TLEN / TTILE)  // 64 chunks per batch
#define EPITCH 258             // padded fp32 LDS row pitch (floats)
#define BPITCH 264             // padded bf16 LDS row pitch (elements)

typedef __attribute__((ext_vector_type(2)))  float  v2f;
typedef __attribute__((ext_vector_type(4)))  float  v4f;
typedef __attribute__((ext_vector_type(8)))  float  v8f;
typedef __attribute__((ext_vector_type(16))) __bf16 v16bf;

#define LOG2E      1.4426950408889634f
#define TWO_LOG2E  2.8853900817779268f

__device__ __forceinline__ float fast_exp(float x) {
  return __builtin_amdgcn_exp2f(x * LOG2E);
}
// tanh(x) = 1 - 2/(exp(2x)+1); saturates correctly at +-1
__device__ __forceinline__ float fast_tanh(float x) {
  float e = __builtin_amdgcn_exp2f(x * TWO_LOG2E);
  return 1.0f - 2.0f * __builtin_amdgcn_rcpf(e + 1.0f);
}

// ---------------------------------------------------------------------------
// Kernel W: one-time w1 (f32) -> bf16 copy in workspace (256 KB, L2-resident)
// ---------------------------------------------------------------------------
__global__ void bahdanau_w1_bf16_kernel(const float* __restrict__ w1,
                                        __bf16* __restrict__ w1b) {
  int i = blockIdx.x * 256 + threadIdx.x;
  w1b[i] = (__bf16)w1[i];
}

// ---------------------------------------------------------------------------
// Kernel 0: hb[b,a] = b1[a]+b2[a]+b3[a] + sum_k h[b,k]*w2[k,a] + c[b,k]*w3[k,a]
// bv is dropped: it cancels in softmax over T.
// ---------------------------------------------------------------------------
__global__ void bahdanau_hidden_bias_kernel(
    const float* __restrict__ h, const float* __restrict__ c,
    const float* __restrict__ w2, const float* __restrict__ b2,
    const float* __restrict__ w3, const float* __restrict__ b3,
    const float* __restrict__ b1, float* __restrict__ hb) {
  __shared__ float sh[DATT];
  __shared__ float sc[DATT];
  const int b = blockIdx.x;
  const int tid = threadIdx.x;
  for (int i = tid; i < DATT; i += 256) {
    sh[i] = h[b * DATT + i];
    sc[i] = c[b * DATT + i];
  }
  __syncthreads();
  for (int a = tid; a < DATT; a += 256) {
    float acc = b1[a] + b2[a] + b3[a];
    for (int k = 0; k < DATT; ++k) {
      acc = fmaf(sh[k], w2[k * DATT + a], acc);
      acc = fmaf(sc[k], w3[k * DATT + a], acc);
    }
    hb[b * DATT + a] = acc;
  }
}

// ---------------------------------------------------------------------------
// Kernel 1: fused projection (WMMA bf16 16x16x32) + tanh + score + partial
// softmax + partial fp32 weighted context. One block = (batch b, 128-row chunk).
// encoder_out read from HBM exactly once (non-temporal, fp32 in LDS).
// Each wave owns M=32 rows; all 16 A fragments (K=256, bf16) are register
// resident (128 VGPRs), so the inner loop is one 32B LDS B read per 2 WMMAs.
// ---------------------------------------------------------------------------
__global__ __launch_bounds__(128)
void bahdanau_score_partial_kernel(
    const float* __restrict__ enc,     // [B,T,DENC] f32
    const __bf16* __restrict__ w1b,    // [DENC,DATT] bf16 (ws)
    const float* __restrict__ hb,      // [B,DATT] combined bias (ws)
    const float* __restrict__ wv,      // [DATT]
    float* __restrict__ pm,            // [B*NCHUNK] partial max
    float* __restrict__ ps,            // [B*NCHUNK] partial sumexp
    float* __restrict__ pctx) {        // [B*NCHUNK, DENC] partial context
  __shared__ float  eTile[TTILE * EPITCH];   // 128 x 256 fp32 encoder tile
  __shared__ __bf16 w1B[16 * BPITCH];        // 16 cols of w1 (bf16), [n][k]
  __shared__ float  hbT[DATT];
  __shared__ float  wvT[DATT];
  __shared__ float  sScores[TTILE];
  __shared__ float  sP[TTILE];
  __shared__ float  red[64];

  const int b = blockIdx.y;
  const int chunk = blockIdx.x;
  const int t0 = chunk * TTILE;
  const int tid = threadIdx.x;
  const int lane = tid & 31;
  const int wave = tid >> 5;          // 4 waves
  const int row0 = wave * 32;         // each wave owns 32 rows (2 strips of 16)

  // ---- stage encoder tile (coalesced, non-temporal 16B vectors) ----
  const v4f* enc4 = (const v4f*)(enc + ((size_t)b * TLEN + t0) * DENC);
  for (int i = tid; i < TTILE * (DENC / 4); i += 128) {
    int r = i / (DENC / 4);
    int c4 = i % (DENC / 4);
    v4f v = __builtin_nontemporal_load(&enc4[(size_t)r * (DENC / 4) + c4]);
    float* dst = &eTile[r * EPITCH + c4 * 4];
    dst[0] = v.x; dst[1] = v.y; dst[2] = v.z; dst[3] = v.w;
  }
  for (int i = tid; i < DATT; i += 128) {
    hbT[i] = hb[b * DATT + i];
    wvT[i] = wv[i];
  }
  __syncthreads();

  const int half = lane >> 4;          // lane half: selects K sub-block / C rows
  const int l16 = lane & 15;

  // ---- build all bf16 A fragments in registers: 2 strips x 8 K-chunks ----
  // A operand, contiguous k order chosen here (B matches): lanes 0-15 (half=0) hold K {0..15},
  // lanes 16-31 (half=1) hold K {16..31}; 2 values packed per VGPR.
  v16bf Afrag[2][8];                   // 16 x 8 VGPRs = 128 VGPRs
#pragma unroll
  for (int s = 0; s < 2; ++s) {
    const int row = row0 + s * 16 + l16;
#pragma unroll
    for (int kc = 0; kc < 8; ++kc) {
      const int kbase = kc * 32 + half * 16;
      v16bf a;
#pragma unroll
      for (int v = 0; v < 8; ++v) {
        int K = kbase + 2 * v;
        v2f f = *(const v2f*)(&eTile[row * EPITCH + K]);
        a[2 * v]     = (__bf16)f.x;
        a[2 * v + 1] = (__bf16)f.y;
      }
      Afrag[s][kc] = a;
    }
  }

  float scoreAcc0[8], scoreAcc1[8];
#pragma unroll
  for (int v = 0; v < 8; ++v) { scoreAcc0[v] = 0.0f; scoreAcc1[v] = 0.0f; }

  for (int n = 0; n < DATT / 16; ++n) {   // 32 column tiles of w1
    // stage w1b[:, n*16 .. n*16+15] transposed into w1B[col][k]
    for (int i = tid; i < 16 * DENC; i += 128) {
      int k = i >> 4;
      int c = i & 15;
      w1B[c * BPITCH + k] = w1b[k * DATT + n * 16 + c];
    }
    __syncthreads();

    v8f C0 = {}, C1 = {};
#pragma unroll
    for (int kc = 0; kc < 8; ++kc) {
      // B operand, same contiguous k order as A: lane holds 16 consecutive K for its column;
      // half 0 -> K kc*32..+15, half 1 -> K kc*32+16..+31 (contiguous 32B).
      v16bf Bf = *(const v16bf*)(&w1B[l16 * BPITCH + kc * 32 + half * 16]);
      C0 = __builtin_amdgcn_wmma_f32_16x16x32_bf16(
          false, Afrag[0][kc], false, Bf, (short)0, C0, false, false);
      C1 = __builtin_amdgcn_wmma_f32_16x16x32_bf16(
          false, Afrag[1][kc], false, Bf, (short)0, C1, false, false);
    }

    // per-lane column-slice accumulation (cross-lane reduction deferred)
    float hbv = hbT[n * 16 + l16];
    float wvv = wvT[n * 16 + l16];
#pragma unroll
    for (int v = 0; v < 8; ++v) {
      scoreAcc0[v] = fmaf(fast_tanh(C0[v] + hbv), wvv, scoreAcc0[v]);
      scoreAcc1[v] = fmaf(fast_tanh(C1[v] + hbv), wvv, scoreAcc1[v]);
    }
    __syncthreads();   // before w1B is overwritten
  }

  // ---- one cross-lane reduction over the 16 lanes sharing each row ----
#pragma unroll
  for (int v = 0; v < 8; ++v) {
    float t0v = scoreAcc0[v];
    t0v += __shfl_xor(t0v, 8); t0v += __shfl_xor(t0v, 4);
    t0v += __shfl_xor(t0v, 2); t0v += __shfl_xor(t0v, 1);
    scoreAcc0[v] = t0v;
    float t1v = scoreAcc1[v];
    t1v += __shfl_xor(t1v, 8); t1v += __shfl_xor(t1v, 4);
    t1v += __shfl_xor(t1v, 2); t1v += __shfl_xor(t1v, 1);
    scoreAcc1[v] = t1v;
  }
  if (l16 == 0) {
#pragma unroll
    for (int v = 0; v < 8; ++v) {
      sScores[row0 + half * 8 + v]      = scoreAcc0[v];   // strip 0: C rows v / v+8
      sScores[row0 + 16 + half * 8 + v] = scoreAcc1[v];   // strip 1
    }
  }
  __syncthreads();

  // ---- block-local softmax partials over the 128 rows ----
  if (tid < 64) red[tid] = fmaxf(sScores[tid], sScores[tid + 64]);
  __syncthreads();
  for (int off = 32; off > 0; off >>= 1) {
    if (tid < off) red[tid] = fmaxf(red[tid], red[tid + off]);
    __syncthreads();
  }
  float mblk = red[0];
  __syncthreads();

  sP[tid] = fast_exp(sScores[tid] - mblk);   // 128 threads = 128 rows
  __syncthreads();
  if (tid < 64) red[tid] = sP[tid] + sP[tid + 64];
  __syncthreads();
  for (int off = 32; off > 0; off >>= 1) {
    if (tid < off) red[tid] += red[tid + off];
    __syncthreads();
  }
  float sblk = red[0];

  // ---- partial context (fp32): ctx[d] = sum_i p[i] * e[i][d] ----
  float acc0 = 0.0f, acc1 = 0.0f;
  for (int i = 0; i < TTILE; ++i) {
    float p = sP[i];
    acc0 = fmaf(p, eTile[i * EPITCH + tid], acc0);
    acc1 = fmaf(p, eTile[i * EPITCH + tid + 128], acc1);
  }

  const int idx = b * NCHUNK + chunk;
  pctx[(size_t)idx * DENC + tid]       = acc0;
  pctx[(size_t)idx * DENC + tid + 128] = acc1;
  if (tid == 0) {
    pm[idx] = mblk;
    ps[idx] = sblk;
  }
}

// ---------------------------------------------------------------------------
// Kernel 2: combine NCHUNK partials per batch -> context [B, DENC]
// ---------------------------------------------------------------------------
__global__ void bahdanau_combine_kernel(
    const float* __restrict__ pm, const float* __restrict__ ps,
    const float* __restrict__ pctx, float* __restrict__ out) {
  __shared__ float sm[NCHUNK];
  __shared__ float sfac[NCHUNK];
  __shared__ float red[NCHUNK];
  const int b = blockIdx.x;
  const int tid = threadIdx.x;

  if (tid < NCHUNK) { sm[tid] = pm[b * NCHUNK + tid]; red[tid] = sm[tid]; }
  __syncthreads();
  for (int off = NCHUNK / 2; off > 0; off >>= 1) {
    if (tid < off) red[tid] = fmaxf(red[tid], red[tid + off]);
    __syncthreads();
  }
  float M = red[0];
  __syncthreads();

  if (tid < NCHUNK) {
    float f = fast_exp(sm[tid] - M);
    sfac[tid] = f;
    red[tid] = ps[b * NCHUNK + tid] * f;
  }
  __syncthreads();
  for (int off = NCHUNK / 2; off > 0; off >>= 1) {
    if (tid < off) red[tid] += red[tid + off];
    __syncthreads();
  }
  float S = red[0];
  __syncthreads();

  float acc = 0.0f;
  for (int i = 0; i < NCHUNK; ++i)
    acc = fmaf(pctx[((size_t)b * NCHUNK + i) * DENC + tid], sfac[i], acc);
  out[b * DENC + tid] = acc * __builtin_amdgcn_rcpf(S);
}

// ---------------------------------------------------------------------------
extern "C" void kernel_launch(void* const* d_in, const int* in_sizes, int n_in,
                              void* d_out, int out_size, void* d_ws, size_t ws_size,
                              hipStream_t stream) {
  const float* enc = (const float*)d_in[0];   // [32,8192,256]
  const float* h   = (const float*)d_in[1];   // [32,512]
  const float* c   = (const float*)d_in[2];   // [32,512]
  const float* w1  = (const float*)d_in[3];   // [256,512]
  const float* b1  = (const float*)d_in[4];   // [512]
  const float* w2  = (const float*)d_in[5];   // [512,512]
  const float* b2  = (const float*)d_in[6];   // [512]
  const float* w3  = (const float*)d_in[7];   // [512,512]
  const float* b3  = (const float*)d_in[8];   // [512]
  const float* wv  = (const float*)d_in[9];   // [512,1]
  // d_in[10] = bv: constant shift of all scores -> cancels in softmax.

  float* ws = (float*)d_ws;
  float* hb   = ws;                                   // 32*512
  float* pm   = hb + BATCH * DATT;                    // 32*64
  float* ps   = pm + BATCH * NCHUNK;                  // 32*64
  float* pctx = ps + BATCH * NCHUNK;                  // 32*64*256
  __bf16* w1b = (__bf16*)(pctx + (size_t)BATCH * NCHUNK * DENC); // 256*512 bf16

  float* out = (float*)d_out;                         // [32,256]

  bahdanau_w1_bf16_kernel<<<dim3((DENC * DATT) / 256), dim3(256), 0, stream>>>(
      w1, w1b);

  bahdanau_hidden_bias_kernel<<<dim3(BATCH), dim3(256), 0, stream>>>(
      h, c, w2, b2, w3, b3, b1, hb);

  bahdanau_score_partial_kernel<<<dim3(NCHUNK, BATCH), dim3(128), 0, stream>>>(
      enc, w1b, hb, wv, pm, ps, pctx);

  bahdanau_combine_kernel<<<dim3(BATCH), dim3(DENC), 0, stream>>>(
      pm, ps, pctx, out);

  (void)in_sizes; (void)n_in; (void)out_size; (void)ws_size;
}
